// LSTM_55465207661165
// MI455X (gfx1250) — matrix operand order repaired and checked by reading, not compiled, not run on hardware
//
#include <hip/hip_runtime.h>
#include <hip/hip_bf16.h>

// ---------------------------------------------------------------------------
// LSTM on MI455X (gfx1250): bf16 WMMA (16x16x32, f32 accum) for all GEMMs.
// Step kernel: blocks of 8 waves share one n-tile; the 64xK B chunk is staged
// through LDS (double-buffered) so WT is read from L2 once per 8 waves.
// 256 sequential step launches (h ping-pong), then output projection.
// ---------------------------------------------------------------------------

typedef __attribute__((ext_vector_type(16))) __bf16 v16bf;
typedef __attribute__((ext_vector_type(8)))  __bf16 v8bf;
typedef __attribute__((ext_vector_type(8)))  float  v8f;
typedef __attribute__((ext_vector_type(4)))  float  v4f;

#define BATCH  256
#define SEQ    256
#define N_IN   512
#define HDIM   1024
#define M_OUT  512
#define KTOT   (N_IN + HDIM)   // 1536
#define G4     (4 * HDIM)      // 4096
#define KC     64              // K rows staged per LDS chunk
#define NCHUNK (KTOT / KC)     // 24

__device__ __forceinline__ float sigmoidf_(float x) {
    return 1.0f / (1.0f + __expf(-x));
}

// A-fragment (16x32 bf16, MxK) per CDNA5 layout:
//   lane<16 : elems 0..7 = K(k0+0..7),  elems 8..15 = K(k0+16..23)
//   lane>=16: elems 0..7 = K(k0+8..15), elems 8..15 = K(k0+24..31)
__device__ __forceinline__ v16bf load_a_f32(const float* __restrict__ src, int ld,
                                            int m, int k0, int lane) {
    const int kb = (lane < 16) ? 0 : 8;
    const float* p0 = src + (size_t)m * ld + k0 + kb;
    const float* p1 = p0 + 16;
    v4f x0 = *(const v4f*)(p0);
    v4f x1 = *(const v4f*)(p0 + 4);
    v4f x2 = *(const v4f*)(p1);
    v4f x3 = *(const v4f*)(p1 + 4);
    v16bf a;
#pragma unroll
    for (int e = 0; e < 4; ++e) {
        a[e]      = (__bf16)x0[e];
        a[4 + e]  = (__bf16)x1[e];
        a[8 + e]  = (__bf16)x2[e];
        a[12 + e] = (__bf16)x3[e];
    }
    return a;
}

__device__ __forceinline__ v16bf load_a_bf16(const __bf16* __restrict__ src, int ld,
                                             int m, int k0, int lane) {
    const int kb = (lane < 16) ? 0 : 8;
    const __bf16* p = src + (size_t)m * ld + k0 + kb;
    union { v16bf v; v8bf h[2]; } u;
    u.h[0] = *(const v8bf*)(p);
    u.h[1] = *(const v8bf*)(p + 16);
    return u.v;
}

// ---------------------------------------------------------------------------
// Prep: WT[k][j] = bf16(Wcat[j][k]); Wcat rows = [Wgx|Wix|Wfx|Wox ; W*h].
// ---------------------------------------------------------------------------
__global__ void prep_weights_kernel(const float* __restrict__ Wgx, const float* __restrict__ Wix,
                                    const float* __restrict__ Wfx, const float* __restrict__ Wox,
                                    const float* __restrict__ Wgh, const float* __restrict__ Wih,
                                    const float* __restrict__ Wfh, const float* __restrict__ Woh,
                                    __bf16* __restrict__ WT) {
    int i = blockIdx.x * blockDim.x + threadIdx.x;
    if (i >= KTOT * G4) return;
    int k  = i >> 12;        // / 4096
    int j  = i & 4095;
    int q  = j >> 10;        // gate index
    int jr = j & 1023;
    const float* Wxq = (q == 0) ? Wgx : (q == 1) ? Wix : (q == 2) ? Wfx : Wox;
    const float* Whq = (q == 0) ? Wgh : (q == 1) ? Wih : (q == 2) ? Wfh : Woh;
    float v = (k < N_IN) ? Wxq[jr * N_IN + k] : Whq[jr * HDIM + (k - N_IN)];
    WT[i] = (__bf16)v;
}

__global__ void prep_wp_kernel(const float* __restrict__ Wp, __bf16* __restrict__ WpT) {
    int i = blockIdx.x * blockDim.x + threadIdx.x;
    if (i >= HDIM * M_OUT) return;
    int k = i / M_OUT;
    int j = i - k * M_OUT;
    WpT[i] = (__bf16)Wp[j * HDIM + k];   // WpT[k][j] = Wp[j][k]
}

__global__ void init_state_kernel(__bf16* __restrict__ h0, float* __restrict__ c, int n) {
    int i = blockIdx.x * blockDim.x + threadIdx.x;
    if (i < n) { h0[i] = (__bf16)0.0f; c[i] = 0.0f; }
}

// ---------------------------------------------------------------------------
// One LSTM timestep.
// Grid: 128 blocks x 256 thr. Block -> (ng = n-tile 0..63, mg = 0..1);
// wave w in block -> m-tile = mg*8+w. All 8 waves share the same B columns
// (gate cols n0, n0+H, n0+2H, n0+3H), staged through LDS double buffers:
//   Bs[buf][r*64 + g*16 + n] = WT[kbase + r][g*1024 + n0 + n],  r=0..63
// B fragment: lane L reads col (L&15) of each gate, LDS rows j*32 + 8*(L>>4) + {0..7,16..23}.
// ---------------------------------------------------------------------------
__global__ void __launch_bounds__(256)
lstm_step_kernel(const float*  __restrict__ x,     // [B, S, N_IN] f32
                 const __bf16* __restrict__ WT,    // [KTOT, 4H] bf16, K-major
                 const __bf16* __restrict__ hin,   // [B, H]
                 __bf16*       __restrict__ hout,  // [B, H]
                 float*        __restrict__ cst,   // [B, H]
                 const float*  __restrict__ bg, const float* __restrict__ bi,
                 const float*  __restrict__ bff, const float* __restrict__ bo,
                 int s) {
    __shared__ __align__(32) __bf16 Bs[2][KC * 64];   // 2 x 8 KB

    const int tid  = threadIdx.x;
    const int lane = tid & 31;
    const int w    = tid >> 5;                 // wave in block (0..7)
    const int ng   = blockIdx.x & 63;          // n-tile over HDIM
    const int mg   = blockIdx.x >> 6;          // 0..1
    const int n0   = ng << 4;
    const int m0   = (mg * 8 + w) << 4;
    const int ma   = m0 + (lane & 15);         // A row for this lane

    // stage-copy role for this thread: row r, gate g (one 32B segment)
    const int sr = tid >> 2;                   // 0..63
    const int sg = tid & 3;                    // 0..3
    const __bf16* wsrc = WT + (size_t)sr * G4 + sg * HDIM + n0;
    __bf16* sdst0 = &Bs[0][sr * 64 + sg * 16];
    __bf16* sdst1 = &Bs[1][sr * 64 + sg * 16];

    const float* xs = x + (size_t)s * N_IN;    // A row stride = SEQ*N_IN

    v8f accg = {}, acci = {}, accf = {}, acco = {};

    // prologue: stage chunk 0
    *(v16bf*)sdst0 = *(const v16bf*)(wsrc);
    __syncthreads();

    for (int c = 0; c < NCHUNK; ++c) {
        // stage next chunk into the other buffer while computing this one
        if (c + 1 < NCHUNK) {
            const v16bf nxt = *(const v16bf*)(wsrc + (size_t)(c + 1) * KC * G4);
            *(v16bf*)(((c + 1) & 1) ? sdst1 : sdst0) = nxt;
        }
        const __bf16* bs = Bs[c & 1];
#pragma unroll
        for (int j = 0; j < 2; ++j) {
            const int kglob = c * KC + j * 32;
            v16bf a;
            if (kglob < N_IN)                      // wave-uniform branch
                a = load_a_f32(xs, SEQ * N_IN, ma, kglob, lane);
            else
                a = load_a_bf16(hin, HDIM, ma, kglob - N_IN, lane);
            const int bkb = (lane >> 4) << 3;
            v16bf b0, b1, b2, b3;
#pragma unroll
            for (int e = 0; e < 16; ++e) {
                const int kk = j * 32 + bkb + (e < 8 ? e : e + 8);
                const __bf16* br = bs + kk * 64 + (lane & 15);
                b0[e] = br[0];
                b1[e] = br[16];
                b2[e] = br[32];
                b3[e] = br[48];
            }
            accg = __builtin_amdgcn_wmma_f32_16x16x32_bf16(false, a, false, b0, (short)0, accg, false, false);
            acci = __builtin_amdgcn_wmma_f32_16x16x32_bf16(false, a, false, b1, (short)0, acci, false, false);
            accf = __builtin_amdgcn_wmma_f32_16x16x32_bf16(false, a, false, b2, (short)0, accf, false, false);
            acco = __builtin_amdgcn_wmma_f32_16x16x32_bf16(false, a, false, b3, (short)0, acco, false, false);
        }
        __syncthreads();
    }

    // epilogue: C/D layout -> lane owns col n0+(lane&15), rows rbase..rbase+7
    const int col   = n0 + (lane & 15);
    const int rbase = m0 + ((lane >> 4) << 3);
    const float big = bg[col], bii = bi[col], bfv = bff[col], bov = bo[col];
#pragma unroll
    for (int r = 0; r < 8; ++r) {
        const int m = rbase + r;
        float g = tanhf(accg[r] + big);
        float i = sigmoidf_(acci[r] + bii);
        float f = sigmoidf_(accf[r] + bfv);
        float o = sigmoidf_(acco[r] + bov);
        float cold = cst[(size_t)m * HDIM + col];
        float cnew = g * i + cold * f;
        cst[(size_t)m * HDIM + col] = cnew;
        hout[(size_t)m * HDIM + col] = (__bf16)(tanhf(cnew) * o);
    }
}

// ---------------------------------------------------------------------------
// Final projection: out[256,512] = h @ Wp^T + bp. 512 waves = 64 blocks.
// ---------------------------------------------------------------------------
__global__ void __launch_bounds__(256)
proj_kernel(const __bf16* __restrict__ h, const __bf16* __restrict__ WpT,
            const float* __restrict__ bp, float* __restrict__ out) {
    const int wave = blockIdx.x * (blockDim.x >> 5) + (threadIdx.x >> 5);
    const int lane = threadIdx.x & 31;
    const int nt = wave & 31;            // 32 tiles over M_OUT
    const int mt = wave >> 5;            // 16 tiles over BATCH
    const int m0 = mt << 4;
    const int n0 = nt << 4;
    const int ma = m0 + (lane & 15);

    v8f acc = {};
    for (int k0 = 0; k0 < HDIM; k0 += 32) {
        v16bf a = load_a_bf16(h, HDIM, ma, k0, lane);
        v16bf b;
#pragma unroll
        for (int e = 0; e < 16; ++e)
            b[e] = WpT[(size_t)(k0 + ((lane >> 4) << 3) + (e < 8 ? e : e + 8)) * M_OUT + n0 + (lane & 15)];
        acc = __builtin_amdgcn_wmma_f32_16x16x32_bf16(false, a, false, b, (short)0, acc, false, false);
    }
    const int col   = n0 + (lane & 15);
    const int rbase = m0 + ((lane >> 4) << 3);
    const float bias = bp[col];
#pragma unroll
    for (int r = 0; r < 8; ++r)
        out[(size_t)(rbase + r) * M_OUT + col] = acc[r] + bias;
}

// ---------------------------------------------------------------------------
extern "C" void kernel_launch(void* const* d_in, const int* in_sizes, int n_in,
                              void* d_out, int out_size, void* d_ws, size_t ws_size,
                              hipStream_t stream) {
    (void)in_sizes; (void)n_in; (void)out_size;

    const float* x   = (const float*)d_in[0];
    const float* Wgx = (const float*)d_in[1];
    const float* bg  = (const float*)d_in[2];
    const float* Wgh = (const float*)d_in[3];
    const float* Wix = (const float*)d_in[4];
    const float* bi  = (const float*)d_in[5];
    const float* Wih = (const float*)d_in[6];
    const float* Wfx = (const float*)d_in[7];
    const float* bff = (const float*)d_in[8];
    const float* Wfh = (const float*)d_in[9];
    const float* Wox = (const float*)d_in[10];
    const float* bo  = (const float*)d_in[11];
    const float* Woh = (const float*)d_in[12];
    const float* Wp  = (const float*)d_in[13];
    const float* bp  = (const float*)d_in[14];
    float* out = (float*)d_out;

    char* ws = (char*)d_ws;
    size_t off = 0;
    __bf16* WT  = (__bf16*)(ws + off); off += (size_t)KTOT * G4 * 2;       // 12.6 MB
    __bf16* WpT = (__bf16*)(ws + off); off += (size_t)HDIM * M_OUT * 2;    //  1.0 MB
    __bf16* h0  = (__bf16*)(ws + off); off += (size_t)BATCH * HDIM * 2;    //  0.5 MB
    __bf16* h1  = (__bf16*)(ws + off); off += (size_t)BATCH * HDIM * 2;    //  0.5 MB
    float*  cst = (float*)(ws + off);  off += (size_t)BATCH * HDIM * 4;    //  1.0 MB
    if (ws_size < off) return;  // workspace too small (~15 MB needed)

    prep_weights_kernel<<<(KTOT * G4 + 255) / 256, 256, 0, stream>>>(
        Wgx, Wix, Wfx, Wox, Wgh, Wih, Wfh, Woh, WT);
    prep_wp_kernel<<<(HDIM * M_OUT + 255) / 256, 256, 0, stream>>>(Wp, WpT);
    init_state_kernel<<<(BATCH * HDIM + 255) / 256, 256, 0, stream>>>(h0, cst, BATCH * HDIM);

    // Sequential recurrence: one kernel per step, h ping-pong (read hin, write hout).
    for (int s = 0; s < SEQ; ++s) {
        const __bf16* hin = (s & 1) ? h1 : h0;
        __bf16*      hout = (s & 1) ? h0 : h1;
        lstm_step_kernel<<<128, 256, 0, stream>>>(x, WT, hin, hout, cst,
                                                  bg, bi, bff, bo, s);
    }
    // SEQ even -> final h lives in h0.
    proj_kernel<<<64, 256, 0, stream>>>(h0, WpT, bp, out);
}
